// GPT_81552839016751
// MI455X (gfx1250) — matrix operand order repaired and checked by reading, not compiled, not run on hardware
//
#include <hip/hip_runtime.h>
#include <hip/hip_bf16.h>

// ---------------- problem constants ----------------
static constexpr int Bb = 16, Ss = 128, Vv = 40000, Dd = 768, Hh = 12, HDd = 64, Ff = 3072, Ll = 12;
static constexpr int Mtok = Bb * Ss;           // 2048 token rows

typedef __bf16 bf16_t;
typedef bf16_t v8bf  __attribute__((ext_vector_type(8)));
typedef bf16_t v16bf __attribute__((ext_vector_type(16)));
typedef float  v8f   __attribute__((ext_vector_type(8)));

__device__ __forceinline__ bf16_t f2bf(float f) {
  unsigned u = __builtin_bit_cast(unsigned, f);
  u = (u + 0x7FFFu + ((u >> 16) & 1u)) >> 16;     // round-to-nearest-even
  unsigned short s = (unsigned short)u;
  return __builtin_bit_cast(bf16_t, s);
}

// ---------------- weight cast+transpose ----------------
// dst[l][n*K + k] = (bf16) src[l][ ((n/g)*K + k)*g + (n%g) ]
//   g == N  : plain [K,N] -> [N,K] transpose (Wo, W1, W2, Wout)
//   g == HD : per-head fold [H,D,HD] -> [(h*HD+e), D]      (Wq, Wk, Wv)
__global__ void gpt_cast_transpose(const float* __restrict__ src, bf16_t* __restrict__ dst,
                                   int K, int N, int g, long long sls, long long dls) {
  long long idx = (long long)blockIdx.x * 256 + threadIdx.x;
  if (idx >= (long long)N * K) return;
  int n = (int)(idx / K), k = (int)(idx % K);
  const float* s = src + (long long)blockIdx.y * sls;
  bf16_t*      d = dst + (long long)blockIdx.y * dls;
  d[(long long)n * K + k] = f2bf(s[((long long)(n / g) * K + k) * g + (n % g)]);
}

// ---------------- embedding ----------------
__global__ void gpt_embed(const int* __restrict__ x, const float* __restrict__ bpe,
                          const float* __restrict__ pe, float* __restrict__ h,
                          bf16_t* __restrict__ hb) {
  int row = blockIdx.x;              // b*S + s
  int s   = row % Ss;
  long long tok = x[row];
#pragma unroll
  for (int i = 0; i < 3; i++) {
    int d = threadIdx.x + i * 256;
    float v = bpe[tok * Dd + d] + pe[(long long)s * Dd + d];
    h[(long long)row * Dd + d]  = v;
    hb[(long long)row * Dd + d] = f2bf(v);
  }
}

// ---------------- fragment helpers ----------------
template <int MT>
__device__ __forceinline__ void gpt_ld_a(const bf16_t* const (&arow)[MT], int k0, v16bf (&a)[MT]) {
#pragma unroll
  for (int i = 0; i < MT; i++) {
    v8bf alo = *(const v8bf*)(arow[i] + k0);
    v8bf ahi = *(const v8bf*)(arow[i] + k0 + 16);
    a[i] = __builtin_shufflevector(alo, ahi, 0,1,2,3,4,5,6,7,8,9,10,11,12,13,14,15);
  }
}
__device__ __forceinline__ void gpt_ld_b(const bf16_t* const (&brow)[4], int k0, v16bf (&b)[4]) {
#pragma unroll
  for (int t = 0; t < 4; t++) {
    v8bf blo = *(const v8bf*)(brow[t] + k0);
    v8bf bhi = *(const v8bf*)(brow[t] + k0 + 16);
    b[t] = __builtin_shufflevector(blo, bhi, 0,1,2,3,4,5,6,7,8,9,10,11,12,13,14,15);
  }
}
template <int MT>
__device__ __forceinline__ void gpt_mma(const v16bf (&a)[MT], const v16bf (&b)[4], v8f (&acc)[MT][4]) {
#pragma unroll
  for (int i = 0; i < MT; i++)
#pragma unroll
    for (int t = 0; t < 4; t++)
      acc[i][t] = __builtin_amdgcn_wmma_f32_16x16x32_bf16(
          false, a[i], false, b[t], (short)0, acc[i][t], false, false);
}

// ---------------- generic batched bf16 WMMA GEMM ----------------
// C[z][m][n] = sum_k A[z][m][k] * BT[z][n][k]  (+ bias[n])
// A row-major MxK (row stride lda), BT row-major NxK (row stride ldb).
// MT = M-tiles per wave (wave tile = (16*MT) x 64; block tile = (128*MT) x 64).
// OM = 0: fp32 output to C; OM = 1: bf16 output to Cb.
// Grid: (M/(128*MT), N/64, batch). Exact tiling; K multiple of 32, K >= 64.
template <int MT, int OM>
__global__ void __launch_bounds__(256)
gpt_gemm_bf16(const bf16_t* __restrict__ A, int lda, long long sA,
              const bf16_t* __restrict__ BT, int ldb, long long sB,
              float* __restrict__ C, bf16_t* __restrict__ Cb, int ldc, long long sC,
              int K, const float* __restrict__ bias) {
  A  += (long long)blockIdx.z * sA;
  BT += (long long)blockIdx.z * sB;
  const long long cbase = (long long)blockIdx.z * sC;

  const int wave  = threadIdx.x >> 5;
  const int lane  = threadIdx.x & 31;
  const int mlane = lane & 15;        // row (A) / col (B) within 16
  const int khalf = lane >> 4;        // which K half-group this lane holds
  const int m0 = blockIdx.x * (128 * MT) + wave * (16 * MT);
  const int n0 = blockIdx.y * 64;

  // ISA 16-bit A 16x32 layout: lanes 0-15 hold K {0..7,16..23}, lanes 16-31 {8..15,24..31}.
  const bf16_t* arow[MT];
#pragma unroll
  for (int i = 0; i < MT; i++)
    arow[i] = A + (long long)(m0 + 16 * i + mlane) * lda + khalf * 8;
  // ISA 16-bit B 32x16 layout: lane holds one column; lanes 0-15 K {0..7,16..23}, 16-31 K {8..15,24..31}.
  const bf16_t* brow[4];
#pragma unroll
  for (int t = 0; t < 4; t++)
    brow[t] = BT + (long long)(n0 + 16 * t + mlane) * ldb + khalf * 8;

  v8f acc[MT][4];
#pragma unroll
  for (int i = 0; i < MT; i++)
#pragma unroll
    for (int t = 0; t < 4; t++) acc[i][t] = v8f{};

  // Software-pipelined K loop: fragments for step k0+32 are loaded before the
  // WMMAs of step k0 issue, so v_wmma waits only on the previous step's loads
  // while the next step's loads are in flight.
  v16bf aC[MT], bC[4];
  gpt_ld_a<MT>(arow, 0, aC);
  gpt_ld_b(brow, 0, bC);
#pragma unroll 2
  for (int k0 = 0; k0 < K - 32; k0 += 32) {
    v16bf aN[MT], bN[4];
    gpt_ld_a<MT>(arow, k0 + 32, aN);
    gpt_ld_b(brow, k0 + 32, bN);
    gpt_mma<MT>(aC, bC, acc);
#pragma unroll
    for (int i = 0; i < MT; i++) aC[i] = aN[i];
#pragma unroll
    for (int t = 0; t < 4; t++) bC[t] = bN[t];
  }
  gpt_mma<MT>(aC, bC, acc);   // last K step

  // C/D layout: VGPR r, lanes 0-15 -> (M=r, N=lane); lanes 16-31 -> (M=8+r, N=lane-16).
#pragma unroll
  for (int i = 0; i < MT; i++) {
    const int rbase = m0 + 16 * i + khalf * 8;
#pragma unroll
    for (int t = 0; t < 4; t++) {
      int col = n0 + 16 * t + mlane;
      float badd = bias ? bias[col] : 0.0f;
#pragma unroll
      for (int r = 0; r < 8; r++) {
        float val = acc[i][t][r] + badd;
        long long idx = cbase + (long long)(rbase + r) * ldc + col;
        if (OM == 0) C[idx]  = val;
        else         Cb[idx] = f2bf(val);
      }
    }
  }
}

// ---------------- attention prep: [b,s,h,e] fp32 -> bf16 qb/kb [b,h,s,e], vT [b,h,e,s] ----------------
__global__ void gpt_prep_attn(const float* __restrict__ qf, const float* __restrict__ kf,
                              const float* __restrict__ vf, bf16_t* __restrict__ qb,
                              bf16_t* __restrict__ kb, bf16_t* __restrict__ vT) {
  long long idx = (long long)blockIdx.x * 256 + threadIdx.x;
  if (idx >= (long long)Bb * Hh * Ss * HDd) return;
  int e = (int)(idx % HDd);
  int s = (int)((idx / HDd) % Ss);
  int h = (int)((idx / ((long long)HDd * Ss)) % Hh);
  int b = (int)(idx / ((long long)HDd * Ss * Hh));
  long long src = (((long long)b * Ss + s) * Hh + h) * HDd + e;
  long long dq  = (((long long)b * Hh + h) * Ss + s) * HDd + e;
  long long dv  = (((long long)b * Hh + h) * HDd + e) * Ss + s;
  qb[dq] = f2bf(qf[src]);
  kb[dq] = f2bf(kf[src]);
  vT[dv] = f2bf(vf[src]);
}

// ---------------- masked softmax over rows of att [B,H,S,S]; writes bf16 probs ----------------
__global__ void gpt_softmax_mask(const float* __restrict__ att, const int* __restrict__ ignore,
                                 bf16_t* __restrict__ attb) {
  int row = blockIdx.x;                 // ((b*H+h)*S + q)
  int k   = threadIdx.x;                // 0..127
  int q   = row % Ss;
  int b   = row / (Hh * Ss);
  bool allowed = (k <= q) && (ignore[b * Ss + k] == 0 || k == q);
  float x = allowed ? att[(long long)row * Ss + k] * 0.125f : -3.0e38f;  // 1/sqrt(64)
  __shared__ float red[128];
  red[k] = x; __syncthreads();
  for (int o = 64; o > 0; o >>= 1) { if (k < o) red[k] = fmaxf(red[k], red[k + o]); __syncthreads(); }
  float m = red[0]; __syncthreads();
  float e = allowed ? __expf(x - m) : 0.0f;
  red[k] = e; __syncthreads();
  for (int o = 64; o > 0; o >>= 1) { if (k < o) red[k] += red[k + o]; __syncthreads(); }
  attb[(long long)row * Ss + k] = f2bf(e / red[0]);
}

// ---------------- merge heads: o [b,h,s,e] fp32 -> ob [(b,s),(h,e)] bf16 ----------------
__global__ void gpt_merge_heads(const float* __restrict__ o, bf16_t* __restrict__ ob) {
  long long idx = (long long)blockIdx.x * 256 + threadIdx.x;
  if (idx >= (long long)Bb * Hh * Ss * HDd) return;
  int e = (int)(idx % HDd);
  int s = (int)((idx / HDd) % Ss);
  int h = (int)((idx / ((long long)HDd * Ss)) % Hh);
  int b = (int)(idx / ((long long)HDd * Ss * Hh));
  ob[(((long long)b * Ss + s) * Hh + h) * HDd + e] =
      f2bf(o[(((long long)b * Hh + h) * Ss + s) * HDd + e]);
}

// ---------------- h = LN(h + add); also emits bf16 copy ----------------
__global__ void gpt_residual_ln(float* __restrict__ h, const float* __restrict__ add,
                                const float* __restrict__ w, const float* __restrict__ bias,
                                bf16_t* __restrict__ hb) {
  int row = blockIdx.x, tid = threadIdx.x;
  float v[3]; float s = 0.f, sq = 0.f;
#pragma unroll
  for (int i = 0; i < 3; i++) {
    int d = tid + i * 256;
    float x = h[(long long)row * Dd + d] + add[(long long)row * Dd + d];
    v[i] = x; s += x; sq += x * x;
  }
  __shared__ float rs[256], rq[256];
  rs[tid] = s; rq[tid] = sq; __syncthreads();
  for (int o = 128; o > 0; o >>= 1) { if (tid < o) { rs[tid] += rs[tid + o]; rq[tid] += rq[tid + o]; } __syncthreads(); }
  float mean = rs[0] * (1.0f / Dd);
  float var  = rq[0] * (1.0f / Dd) - mean * mean;
  float r = rsqrtf(var + 1e-5f);
#pragma unroll
  for (int i = 0; i < 3; i++) {
    int d = tid + i * 256;
    float y = (v[i] - mean) * r * w[d] + bias[d];
    h[(long long)row * Dd + d]  = y;
    hb[(long long)row * Dd + d] = f2bf(y);
  }
}

// ---------------- h = LN(h + gelu(f)); exact erf GELU (applied after W2, per reference) ----------------
__global__ void gpt_gelu_residual_ln(float* __restrict__ h, const float* __restrict__ f,
                                     const float* __restrict__ w, const float* __restrict__ bias,
                                     bf16_t* __restrict__ hb) {
  int row = blockIdx.x, tid = threadIdx.x;
  float v[3]; float s = 0.f, sq = 0.f;
#pragma unroll
  for (int i = 0; i < 3; i++) {
    int d = tid + i * 256;
    float t = f[(long long)row * Dd + d];
    float g = 0.5f * t * (1.0f + erff(t * 0.70710678118654752f));
    float x = h[(long long)row * Dd + d] + g;
    v[i] = x; s += x; sq += x * x;
  }
  __shared__ float rs[256], rq[256];
  rs[tid] = s; rq[tid] = sq; __syncthreads();
  for (int o = 128; o > 0; o >>= 1) { if (tid < o) { rs[tid] += rs[tid + o]; rq[tid] += rq[tid + o]; } __syncthreads(); }
  float mean = rs[0] * (1.0f / Dd);
  float var  = rq[0] * (1.0f / Dd) - mean * mean;
  float r = rsqrtf(var + 1e-5f);
#pragma unroll
  for (int i = 0; i < 3; i++) {
    int d = tid + i * 256;
    float y = (v[i] - mean) * r * w[d] + bias[d];
    h[(long long)row * Dd + d]  = y;
    hb[(long long)row * Dd + d] = f2bf(y);
  }
}

// =========================================================================
extern "C" void kernel_launch(void* const* d_in, const int* in_sizes, int n_in,
                              void* d_out, int out_size, void* d_ws, size_t ws_size,
                              hipStream_t stream) {
  (void)in_sizes; (void)n_in; (void)out_size; (void)ws_size;
  const int*   x      = (const int*)  d_in[0];
  const int*   ignore = (const int*)  d_in[1];
  const float* bpe    = (const float*)d_in[2];
  const float* pe     = (const float*)d_in[3];
  const float* Wq     = (const float*)d_in[4];
  const float* bq     = (const float*)d_in[5];
  const float* Wk     = (const float*)d_in[6];
  const float* bk     = (const float*)d_in[7];
  const float* Wv     = (const float*)d_in[8];
  const float* bv     = (const float*)d_in[9];
  const float* Wo     = (const float*)d_in[10];
  const float* bo     = (const float*)d_in[11];
  const float* W1     = (const float*)d_in[12];
  const float* b1     = (const float*)d_in[13];
  const float* W2     = (const float*)d_in[14];
  const float* b2     = (const float*)d_in[15];
  const float* ln1w   = (const float*)d_in[16];
  const float* ln1b   = (const float*)d_in[17];
  const float* ln2w   = (const float*)d_in[18];
  const float* ln2b   = (const float*)d_in[19];
  const float* Wout   = (const float*)d_in[20];
  const float* bout   = (const float*)d_in[21];
  float* out = (float*)d_out;

  // ---- workspace bump allocator (256-B aligned) ----
  char* ws = (char*)d_ws;
  size_t off = 0;
  auto alloc = [&](size_t bytes) -> char* {
    char* p = ws + off;
    off = (off + bytes + 255) & ~(size_t)255;
    return p;
  };
  bf16_t* WqT   = (bf16_t*)alloc((size_t)Ll * Dd * Dd * 2);
  bf16_t* WkT   = (bf16_t*)alloc((size_t)Ll * Dd * Dd * 2);
  bf16_t* WvT   = (bf16_t*)alloc((size_t)Ll * Dd * Dd * 2);
  bf16_t* WoT   = (bf16_t*)alloc((size_t)Ll * Dd * Dd * 2);
  bf16_t* W1T   = (bf16_t*)alloc((size_t)Ll * Ff * Dd * 2);
  bf16_t* W2T   = (bf16_t*)alloc((size_t)Ll * Dd * Ff * 2);
  bf16_t* WoutT = (bf16_t*)alloc((size_t)Vv * Dd * 2);
  float*  h     = (float*) alloc((size_t)Mtok * Dd * 4);
  bf16_t* hb    = (bf16_t*)alloc((size_t)Mtok * Dd * 2);
  float*  qf    = (float*) alloc((size_t)Mtok * Dd * 4);
  float*  kf    = (float*) alloc((size_t)Mtok * Dd * 4);
  float*  vf    = (float*) alloc((size_t)Mtok * Dd * 4);
  bf16_t* qb    = (bf16_t*)alloc((size_t)Mtok * Dd * 2);
  bf16_t* kb    = (bf16_t*)alloc((size_t)Mtok * Dd * 2);
  bf16_t* vT    = (bf16_t*)alloc((size_t)Mtok * Dd * 2);
  float*  att   = (float*) alloc((size_t)Bb * Hh * Ss * Ss * 4);
  bf16_t* attb  = (bf16_t*)alloc((size_t)Bb * Hh * Ss * Ss * 2);
  float*  o     = (float*) alloc((size_t)Mtok * Dd * 4);
  bf16_t* ob    = (bf16_t*)alloc((size_t)Mtok * Dd * 2);
  float*  tmp   = (float*) alloc((size_t)Mtok * Dd * 4);
  bf16_t* f1b   = (bf16_t*)alloc((size_t)Mtok * Ff * 2);
  float*  f2    = (float*) alloc((size_t)Mtok * Dd * 4);

  // ---- weight prep: cast + transpose all GEMM weights to bf16 [N][K] ----
  {
    int nk = Dd * Dd;
    dim3 g((nk + 255) / 256, Ll);
    gpt_cast_transpose<<<g, 256, 0, stream>>>(Wq, WqT, Dd, Dd, HDd, (long long)Hh * Dd * HDd, (long long)Dd * Dd);
    gpt_cast_transpose<<<g, 256, 0, stream>>>(Wk, WkT, Dd, Dd, HDd, (long long)Hh * Dd * HDd, (long long)Dd * Dd);
    gpt_cast_transpose<<<g, 256, 0, stream>>>(Wv, WvT, Dd, Dd, HDd, (long long)Hh * Dd * HDd, (long long)Dd * Dd);
    gpt_cast_transpose<<<g, 256, 0, stream>>>(Wo, WoT, Dd, Dd, Dd, (long long)Dd * Dd, (long long)Dd * Dd);
    int nk1 = Ff * Dd;
    gpt_cast_transpose<<<dim3((nk1 + 255) / 256, Ll), 256, 0, stream>>>(W1, W1T, Dd, Ff, Ff, (long long)Dd * Ff, (long long)Ff * Dd);
    gpt_cast_transpose<<<dim3((nk1 + 255) / 256, Ll), 256, 0, stream>>>(W2, W2T, Ff, Dd, Dd, (long long)Ff * Dd, (long long)Dd * Ff);
    long long nko = (long long)Vv * Dd;
    gpt_cast_transpose<<<dim3((unsigned)((nko + 255) / 256), 1), 256, 0, stream>>>(Wout, WoutT, Dd, Vv, Vv, 0, 0);
  }

  // MT=1: block tile 128x64 (small / batched GEMMs).  MT=2: block tile 256x64 (heavy GEMMs).
  auto gemm_f32 = [&](const bf16_t* A, int lda, long long sA, const bf16_t* BT, int ldb, long long sB,
                      float* C, int ldc, long long sC, int M, int N, int K,
                      const float* bias, int batch) {
    dim3 grid(M / 128, N / 64, batch);
    gpt_gemm_bf16<1, 0><<<grid, 256, 0, stream>>>(A, lda, sA, BT, ldb, sB, C, nullptr, ldc, sC, K, bias);
  };
  auto gemm2_f32 = [&](const bf16_t* A, int lda, const bf16_t* BT, int ldb,
                       float* C, int ldc, int M, int N, int K, const float* bias) {
    dim3 grid(M / 256, N / 64, 1);
    gpt_gemm_bf16<2, 0><<<grid, 256, 0, stream>>>(A, lda, 0, BT, ldb, 0, C, nullptr, ldc, 0, K, bias);
  };
  auto gemm2_bf16 = [&](const bf16_t* A, int lda, const bf16_t* BT, int ldb,
                        bf16_t* Cb, int ldc, int M, int N, int K, const float* bias) {
    dim3 grid(M / 256, N / 64, 1);
    gpt_gemm_bf16<2, 1><<<grid, 256, 0, stream>>>(A, lda, 0, BT, ldb, 0, nullptr, Cb, ldc, 0, K, bias);
  };

  // ---- embedding ----
  gpt_embed<<<Mtok, 256, 0, stream>>>(x, bpe, pe, h, hb);

  const int bhse = Bb * Hh * Ss * HDd;
  for (int l = 0; l < Ll; l++) {
    // QKV projections: [2048,768] x [768,768]^T -> [b,s,h,e]
    gemm_f32(hb, Dd, 0, WqT + (size_t)l * Dd * Dd, Dd, 0, qf, Dd, 0, Mtok, Dd, Dd, bq + (size_t)l * Dd, 1);
    gemm_f32(hb, Dd, 0, WkT + (size_t)l * Dd * Dd, Dd, 0, kf, Dd, 0, Mtok, Dd, Dd, bk + (size_t)l * Dd, 1);
    gemm_f32(hb, Dd, 0, WvT + (size_t)l * Dd * Dd, Dd, 0, vf, Dd, 0, Mtok, Dd, Dd, bv + (size_t)l * Dd, 1);
    gpt_prep_attn<<<(bhse + 255) / 256, 256, 0, stream>>>(qf, kf, vf, qb, kb, vT);
    // scores: per (b,h): q[S,HD] x k[S,HD]^T -> att[S,S]
    gemm_f32(qb, HDd, (long long)Ss * HDd, kb, HDd, (long long)Ss * HDd,
             att, Ss, (long long)Ss * Ss, Ss, Ss, HDd, nullptr, Bb * Hh);
    gpt_softmax_mask<<<Bb * Hh * Ss, 128, 0, stream>>>(att, ignore, attb);
    // o = att @ v: per (b,h): [S,S] x vT[HD,S]^T -> o[S,HD]
    gemm_f32(attb, Ss, (long long)Ss * Ss, vT, Ss, (long long)HDd * Ss,
             o, HDd, (long long)Ss * HDd, Ss, HDd, Ss, nullptr, Bb * Hh);
    gpt_merge_heads<<<(bhse + 255) / 256, 256, 0, stream>>>(o, ob);
    // output projection + residual + LN1
    gemm_f32(ob, Dd, 0, WoT + (size_t)l * Dd * Dd, Dd, 0, tmp, Dd, 0, Mtok, Dd, Dd, bo + (size_t)l * Dd, 1);
    gpt_residual_ln<<<Mtok, 256, 0, stream>>>(h, tmp, ln1w + (size_t)l * Dd, ln1b + (size_t)l * Dd, hb);
    // MLP: f = gelu((h@W1+b1)@W2+b2); h = LN2(h + f).  GEMM1 writes bf16 directly.
    gemm2_bf16(hb, Dd, W1T + (size_t)l * Ff * Dd, Dd, f1b, Ff, Mtok, Ff, Dd, b1 + (size_t)l * Ff);
    gemm2_f32(f1b, Ff, W2T + (size_t)l * Dd * Ff, Ff, f2, Dd, Mtok, Dd, Ff, b2 + (size_t)l * Dd);
    gpt_gelu_residual_ln<<<Mtok, 256, 0, stream>>>(h, f2, ln2w + (size_t)l * Dd, ln2b + (size_t)l * Dd, hb);
  }

  // ---- final vocab projection: [2048,768] x [768,40000] -> d_out ----
  gemm2_f32(hb, Dd, WoutT, Dd, out, Vv, Mtok, Vv, Dd, bout);
}
